// WRFInfo_80238579024455
// MI455X (gfx1250) — compile-verified, matrix operand order checked by reading, not run on hardware
//
#include <hip/hip_runtime.h>
#include <hip/hip_bf16.h>

// ---------------- problem constants ----------------
#define T_STEPS 12
#define BATCH   8
#define CIN_X   64      // wrf channels
#define HID     32
#define CIN     96      // CIN_X + HID
#define COUT    128     // 4*HID
#define HW      64
#define PIX     4096    // HW*HW
#define PADW    68      // HW + 4 (pad 2 each side for 5x5 SAME)
#define PADPIX  4624    // 68*68
#define KDIM    2400    // 25*CIN, ordered k = (ky*5+kx)*96 + ci
#define NCHUNK  75      // KDIM / 32

// GEMM tiling
#define MT   128
#define KB   32
#define LDA  40   // bf16 elems per LDS row (40*2=80 bytes, rows stay 16B aligned)
#define LDB  40

typedef __attribute__((ext_vector_type(16))) __bf16 v16bf;
typedef __attribute__((ext_vector_type(8)))  __bf16 v8bf;
typedef __attribute__((ext_vector_type(8)))  float  v8f;

// ---------------- helper kernels ----------------

__global__ void zero_f32(float* __restrict__ p, int n) {
    int i = blockIdx.x * 256 + threadIdx.x;
    if (i < n) p[i] = 0.0f;
}

// out[0][b][32+hid][yx] = 0 (reverse direction never writes t=0)
__global__ void zero_rev0(float* __restrict__ out) {
    int tid = blockIdx.x * 256 + threadIdx.x;       // B*HID*PIX threads
    int yx  = tid & (PIX - 1);
    int r   = tid >> 12;
    int hid = r & (HID - 1);
    int b   = r >> 5;
    out[(((b * 64) + 32 + hid) << 12) + yx] = 0.0f;
}

// transpose W [n][ci][ky][kx] (fp32) -> wp [n][kpos][ci] (bf16), kpos = ky*5+kx
__global__ void convert_w(const float* __restrict__ w, __bf16* __restrict__ wp, int n) {
    int i = blockIdx.x * 256 + threadIdx.x;
    if (i < n) {
        int cout = i / KDIM;
        int k    = i % KDIM;
        int kpos = k / CIN;
        int ci   = k % CIN;
        wp[i] = (__bf16)w[cout * KDIM + ci * 25 + kpos];
    }
}

// Build padded bf16 NHWC buffer [B][68][68][96] = concat(x_t, h);
// also emit pre-update h into out[t].
__global__ void pack_xh(const float* __restrict__ wrf, const float* __restrict__ h,
                        __bf16* __restrict__ xh, float* __restrict__ out,
                        int t, int dirOff) {
    int tid = blockIdx.x * 256 + threadIdx.x;       // B*4624*96 threads (exact)
    int ci = tid % CIN;
    int r  = tid / CIN;
    int p  = r % PADPIX;
    int b  = r / PADPIX;
    int py = p / PADW, px = p % PADW;
    int y  = py - 2,   x  = px - 2;
    float val = 0.0f;
    if (y >= 0 && y < HW && x >= 0 && x < HW) {
        int yx = (y << 6) + x;
        if (ci < CIN_X) {
            val = wrf[(((t * BATCH + b) * 64 + ci) << 12) + yx];
        } else {
            val = h[((b * HID) + (ci - CIN_X)) * PIX + yx];
            // pre-update hidden state goes to output slot t
            out[(((t * BATCH + b) * 64 + dirOff + (ci - CIN_X)) << 12) + yx] = val;
        }
    }
    xh[tid] = (__bf16)val;   // NHWC: ((b*4624 + p)*96 + ci)
}

// uniform (scalar) element offset of K-chunk q within the NHWC padded buffer
__device__ __forceinline__ int chunk_koff(int q) {
    int kpos = q / 3;                 // kernel position 0..24
    int ci0  = (q % 3) << 5;          // 0,32,64
    int ky = kpos / 5, kx = kpos % 5;
    return (ky * PADW + kx) * CIN + ci0;
}

// 32-byte async copy global -> LDS (two b128s; IOFFSET applies to both addresses)
__device__ __forceinline__ void async_copy32(unsigned lds_addr, const __bf16* gptr) {
    unsigned long long ga = (unsigned long long)gptr;
    asm volatile("global_load_async_to_lds_b128 %0, %1, off"
                 :: "v"(lds_addr), "v"(ga) : "memory");
    asm volatile("global_load_async_to_lds_b128 %0, %1, off offset:16"
                 :: "v"(lds_addr), "v"(ga) : "memory");
}

__device__ __forceinline__ void wait_async0() {
    asm volatile("s_wait_asynccnt 0x0" ::: "memory");
}

// ---------------- implicit-GEMM conv via WMMA ----------------
// C[M=32768, N=128] = A[M, K=2400] * B[K, N]; A = im2col(xh NHWC), B = wp [n][k].
// Block: 256 threads = 8 waves, tile 128(M) x 128(N); wave owns 64x32.
// Double-buffered LDS fed by GLOBAL_LOAD_ASYNC_TO_LDS_B128, one barrier per chunk.
__global__ __launch_bounds__(256) void gemm_conv(const __bf16* __restrict__ xh,
                                                 const __bf16* __restrict__ wp,
                                                 float* __restrict__ z) {
    __shared__ __bf16 sA[2][MT * LDA];
    __shared__ __bf16 sB[2][COUT * LDB];

    const int tid  = threadIdx.x;
    const int lane = tid & 31;
    const int wave = tid >> 5;
    const int wm   = (wave >> 2) * 64;    // wave M offset: 0 or 64
    const int wn   = (wave & 3) * 32;     // wave N offset: 0,32,64,96
    const int m0   = blockIdx.x * MT;

    // cooperative loader coords: thread owns one 16-elem half-row of each tile
    const int am = tid >> 1;              // tile row 0..127
    const int ah = tid & 1;               // K-half (0/1)
    const int gm = m0 + am;
    const int ab = gm >> 12;              // batch
    const int ayx = gm & (PIX - 1);
    const int ay = ayx >> 6;
    const int ax = ayx & 63;
    const __bf16* aBase = xh + (size_t)(ab * PADPIX + ay * PADW + ax) * CIN + ah * 16;
    const __bf16* bBase = wp + (size_t)am * KDIM + ah * 16;

    // LDS byte addresses (flat-pointer low 32 bits == LDS offset)
    const unsigned awLds0 = (unsigned)(unsigned long long)(void*)&sA[0][am * LDA + ah * 16];
    const unsigned awLds1 = (unsigned)(unsigned long long)(void*)&sA[1][am * LDA + ah * 16];
    const unsigned bwLds0 = (unsigned)(unsigned long long)(void*)&sB[0][am * LDB + ah * 16];
    const unsigned bwLds1 = (unsigned)(unsigned long long)(void*)&sB[1][am * LDB + ah * 16];

    const v8f vzero = {};
    v8f acc[4][2];
#pragma unroll
    for (int i = 0; i < 4; ++i)
#pragma unroll
        for (int j = 0; j < 2; ++j) acc[i][j] = vzero;

    // chunk 0 -> buffer 0 (async)
    async_copy32(awLds0, aBase);
    async_copy32(bwLds0, bBase);
    wait_async0();
    __syncthreads();

    const int mr = lane & 15;
    const int kb = (lane < 16) ? 0 : 8;   // low lanes: K 0..7 & 16..23; high: 8..15 & 24..31

    for (int it = 0; it < NCHUNK; ++it) {
        const int cur = it & 1;
        if (it < NCHUNK - 1) {            // async-stage next chunk into other buffer
            const unsigned awN = (cur ^ 1) ? awLds1 : awLds0;
            const unsigned bwN = (cur ^ 1) ? bwLds1 : bwLds0;
            async_copy32(awN, aBase + chunk_koff(it + 1));
            async_copy32(bwN, bBase + (it + 1) * KB);
            if (it < NCHUNK - 2)
                __builtin_prefetch(bBase + (it + 2) * KB, 0, 0);  // global_prefetch_b8
        }

        // ---- fragment loads per CDNA5 16-bit A/B VGPR layout ----
        v16bf afr[4], bfr[2];
#pragma unroll
        for (int fm = 0; fm < 4; ++fm) {
            const __bf16* p = &sA[cur][(wm + fm * 16 + mr) * LDA + kb];
            v8bf lo = *(const v8bf*)p;
            v8bf hi = *(const v8bf*)(p + 16);
            afr[fm] = __builtin_shufflevector(lo, hi, 0,1,2,3,4,5,6,7,8,9,10,11,12,13,14,15);
        }
#pragma unroll
        for (int fn = 0; fn < 2; ++fn) {
            const __bf16* p = &sB[cur][(wn + fn * 16 + mr) * LDB + kb];
            v8bf lo = *(const v8bf*)p;
            v8bf hi = *(const v8bf*)(p + 16);
            bfr[fn] = __builtin_shufflevector(lo, hi, 0,1,2,3,4,5,6,7,8,9,10,11,12,13,14,15);
        }
        // ---- 8 WMMAs per K-chunk ----
#pragma unroll
        for (int fm = 0; fm < 4; ++fm)
#pragma unroll
            for (int fn = 0; fn < 2; ++fn)
                acc[fm][fn] = __builtin_amdgcn_wmma_f32_16x16x32_bf16(
                    false, afr[fm], false, bfr[fn], (short)0, acc[fm][fn], false, false);

        if (it < NCHUNK - 1) {            // make staged chunk visible to all waves
            wait_async0();
            __syncthreads();
        }
    }

    // ---- store: D layout lane = col N (mod 16), VGPR v = row M (+8 for high lanes) ----
    const int rofs = (lane < 16) ? 0 : 8;
#pragma unroll
    for (int fm = 0; fm < 4; ++fm) {
#pragma unroll
        for (int fn = 0; fn < 2; ++fn) {
            const int n = wn + fn * 16 + mr;
            const int m = m0 + wm + fm * 16 + rofs;     // 8-aligned, b constant over v
            const int b  = m >> 12;
            const int yx = m & (PIX - 1);
            float* zp = z + ((((b << 7) + n) << 12) + yx);
            const v8f a = acc[fm][fn];
            float4 v0 = make_float4(a[0], a[1], a[2], a[3]);
            float4 v1 = make_float4(a[4], a[5], a[6], a[7]);
            *(float4*)zp       = v0;
            *(float4*)(zp + 4) = v1;
        }
    }
}

// ---------------- peephole LSTM gate update ----------------
__global__ void gates(const float* __restrict__ z, const float* __restrict__ bias,
                      const float* __restrict__ wci, const float* __restrict__ wcf,
                      const float* __restrict__ wco,
                      float* __restrict__ h, float* __restrict__ c) {
    int tid = blockIdx.x * 256 + threadIdx.x;   // B*HID*PIX threads (exact)
    int yx  = tid & (PIX - 1);
    int r   = tid >> 12;
    int hid = r & (HID - 1);
    int b   = r >> 5;
    int pidx = (hid << 12) + yx;
    const float* zb = z + ((size_t)b << 19);    // b * 128 * 4096
    float zi = zb[((      hid) << 12) + yx] + bias[      hid];
    float zf = zb[((32  + hid) << 12) + yx] + bias[32  + hid];
    float zg = zb[((64  + hid) << 12) + yx] + bias[64  + hid];
    float zo = zb[((96  + hid) << 12) + yx] + bias[96  + hid];
    float cv = c[tid];
    float gi = 1.0f / (1.0f + __expf(-(zi + wci[pidx] * cv)));
    float gf = 1.0f / (1.0f + __expf(-(zf + wcf[pidx] * cv)));
    float cn = gf * cv + gi * tanhf(zg);
    float go = 1.0f / (1.0f + __expf(-(zo + wco[pidx] * cn)));
    h[tid] = go * tanhf(cn);
    c[tid] = cn;
}

// ---------------- host orchestration ----------------
extern "C" void kernel_launch(void* const* d_in, const int* in_sizes, int n_in,
                              void* d_out, int out_size, void* d_ws, size_t ws_size,
                              hipStream_t stream) {
    (void)in_sizes; (void)n_in; (void)out_size; (void)ws_size;
    const float* wrf   = (const float*)d_in[0];
    const float* W_f   = (const float*)d_in[1];
    const float* b_f   = (const float*)d_in[2];
    const float* wci_f = (const float*)d_in[3];
    const float* wcf_f = (const float*)d_in[4];
    const float* wco_f = (const float*)d_in[5];
    const float* W_r   = (const float*)d_in[6];
    const float* b_r   = (const float*)d_in[7];
    const float* wci_r = (const float*)d_in[8];
    const float* wcf_r = (const float*)d_in[9];
    const float* wco_r = (const float*)d_in[10];
    float* out = (float*)d_out;

    // workspace carve-up
    char* ws = (char*)d_ws;
    const size_t XH_B = (size_t)BATCH * CIN * PADPIX * 2;          // 7,102,464 -> pad
    const size_t XH_PAD = (XH_B + 255) & ~(size_t)255;
    const size_t Z_B  = (size_t)BATCH * COUT * PIX * 4;            // 16,777,216
    const size_t HC_B = (size_t)BATCH * HID * PIX * 4;             // 4,194,304
    __bf16* xh = (__bf16*)ws;
    float*  z  = (float*)(ws + XH_PAD);
    float*  h  = (float*)(ws + XH_PAD + Z_B);
    float*  c  = (float*)(ws + XH_PAD + Z_B + HC_B);
    __bf16* wp = (__bf16*)(ws + XH_PAD + Z_B + 2 * HC_B);

    const int NHC   = BATCH * HID * PIX;             // 1,048,576
    const int GPACK = (BATCH * CIN * PADPIX) / 256;  // 13872 exact
    const int GGATE = NHC / 256;                     // 4096 exact
    const int GGEMM = (BATCH * PIX) / MT;            // 256

    for (int dir = 0; dir < 2; ++dir) {
        const float* W    = dir ? W_r   : W_f;
        const float* bias = dir ? b_r   : b_f;
        const float* wci  = dir ? wci_r : wci_f;
        const float* wcf  = dir ? wcf_r : wcf_f;
        const float* wco  = dir ? wco_r : wco_f;
        const int dirOff  = dir ? HID : 0;

        zero_f32<<<GGATE, 256, 0, stream>>>(h, NHC);
        zero_f32<<<GGATE, 256, 0, stream>>>(c, NHC);
        convert_w<<<(COUT * KDIM) / 256, 256, 0, stream>>>(W, wp, COUT * KDIM);
        if (dir == 1) zero_rev0<<<GGATE, 256, 0, stream>>>(out);

        if (dir == 0) {
            for (int t = 0; t < T_STEPS; ++t) {
                pack_xh<<<GPACK, 256, 0, stream>>>(wrf, h, xh, out, t, dirOff);
                if (t < T_STEPS - 1) {
                    gemm_conv<<<GGEMM, 256, 0, stream>>>(xh, wp, z);
                    gates<<<GGATE, 256, 0, stream>>>(z, bias, wci, wcf, wco, h, c);
                }
            }
        } else {
            for (int s = T_STEPS - 1; s >= 1; --s) {
                pack_xh<<<GPACK, 256, 0, stream>>>(wrf, h, xh, out, s, dirOff);
                if (s > 1) {
                    gemm_conv<<<GGEMM, 256, 0, stream>>>(xh, wp, z);
                    gates<<<GGATE, 256, 0, stream>>>(z, bias, wci, wcf, wco, h, c);
                }
            }
        }
    }
}
